// VarianceAdaptor_53498112639061
// MI455X (gfx1250) — matrix operand order repaired and checked by reading, not compiled, not run on hardware
//
#include <hip/hip_runtime.h>
#include <hip/hip_bf16.h>

typedef _Float16 v16h __attribute__((ext_vector_type(16)));
typedef _Float16 v8h  __attribute__((ext_vector_type(8)));
typedef float    v8f  __attribute__((ext_vector_type(8)));

#define BATCH 32
#define TT    512
#define LMAX  2048
#define HD    512
#define KW    3
#define NLAY  2
#define NBINS 256
#define NCK   16               // channel chunks of 32 (512/32)
#define BM    128              // block M tile (8 waves x 16 rows)
#define BN    128              // block N tile (8 WMMA subtiles per wave)
#define NROW  130              // staged rows: t-1 .. t+128
#define LDA   40               // padded LDS row stride (halves)
#define CHUNK_H   (32 * 32 * 16)          // halves per (kc) chunk = 16384
#define TAP_H     ((size_t)NCK * CHUNK_H) // halves per tap = 262144
#define WT_LAYER  ((size_t)KW * TAP_H)    // 786432 halves per layer
#define EPSF  1e-5f

struct Frag { v8h h0, h1; };

__device__ __forceinline__ void load_group(Frag* f, const _Float16* wb) {
#pragma unroll
  for (int s = 0; s < 4; ++s) {
    f[s].h0 = *(const v8h*)(wb + s * 512);
    f[s].h1 = *(const v8h*)(wb + s * 512 + 8);
  }
}

__device__ __forceinline__ v16h load_afrag(const _Float16* base) {
  union { v16h v; v8h h[2]; } u;
  u.h[0] = *(const v8h*)base;
  u.h[1] = *(const v8h*)(base + 16);
  return u.v;
}

__device__ __forceinline__ void wmma4(v8f* acc, v16h a, const Frag* B) {
#pragma unroll
  for (int s = 0; s < 4; ++s) {
    union { v16h v; v8h h[2]; } u;
    u.h[0] = B[s].h0; u.h[1] = B[s].h1;
    acc[s] = __builtin_amdgcn_wmma_f32_16x16x32_f16(
        false, a, false, u.v, (short)0, acc[s], false, false);
  }
}

// ---------------------------------------------------------------------------
// Convert f32 activations to f16 (once for x; LN writes f16 directly).
// ---------------------------------------------------------------------------
__global__ __launch_bounds__(256) void f32_to_f16_kernel(
    const float* __restrict__ in, _Float16* __restrict__ out) {
  size_t i = ((size_t)blockIdx.x * 256 + threadIdx.x) * 4;
  float4 v = *(const float4*)(in + i);
  _Float16* o = out + i;
  o[0] = (_Float16)v.x; o[1] = (_Float16)v.y;
  o[2] = (_Float16)v.z; o[3] = (_Float16)v.w;
}

// ---------------------------------------------------------------------------
// Repack conv weights (lay,O,I,K) f32 into per-lane WMMA B-fragment order:
// wt[kcg][n16][lane][j]; kcg = tap*16 + kchunk; K-in-chunk = (j>>3)*16+(lane>>4)*8+(j&7);
// N = n16*16 + (lane&15).
// ---------------------------------------------------------------------------
__global__ __launch_bounds__(256) void repack_w_kernel(
    const float* __restrict__ cw, _Float16* __restrict__ wt) {
  const int lay = blockIdx.y;
  const int idx = blockIdx.x * 256 + threadIdx.x;    // < 786432
  const int j    = idx & 15;
  const int lane = (idx >> 4) & 31;
  const int n16  = (idx >> 9) & 31;
  const int kcg  = idx >> 14;                        // 0..47
  const int tap  = kcg >> 4;
  const int kch  = kcg & 15;
  const int kl   = ((j >> 3) << 4) + ((lane >> 4) << 3) + (j & 7); // K within 32-chunk
  const int chan = (kch << 5) + kl;
  const int o    = (n16 << 4) + (lane & 15);
  wt[(size_t)lay * WT_LAYER + idx] =
      (_Float16)cw[(((size_t)lay * HD + o) * HD + chan) * KW + tap];
}

// ---------------------------------------------------------------------------
// Conv1d(k=3, same) as WMMA GEMM with tap-shared A staging and a
// software-pipelined (2-buffer) B-fragment stream.
// Block: 128M x 128N, 8 waves; 24 WMMAs per wave per barrier.
// ---------------------------------------------------------------------------
__global__ __launch_bounds__(256) void conv_gemm_kernel(
    const _Float16* __restrict__ in_h, const _Float16* __restrict__ wt,
    const float* __restrict__ bias, float* __restrict__ out) {
  __shared__ _Float16 As[2][NROW * LDA];

  const int mBase = blockIdx.x * BM;
  const int nTile = blockIdx.y;
  const int tid   = threadIdx.x;
  const int lane  = tid & 31;
  const int wave  = tid >> 5;
  const int l15   = lane & 15;
  const int kh    = lane >> 4;
  const int mstripe = wave << 4;
  const int bb    = mBase / TT;
  const int tBase = mBase % TT;

  // ---- staging assignment: 260 segments of 16 halves (130 rows x 2) ----
  const int seg0 = tid;
  const int row0 = seg0 >> 1, off0 = (seg0 & 1) << 4;
  const int t_0  = tBase + row0 - 1;
  const bool ok0 = (unsigned)t_0 < (unsigned)TT;
  const _Float16* src0 = in_h + (size_t)(bb * TT + t_0) * HD + off0;
  const int seg1 = tid + 256;
  const int row1 = seg1 >> 1, off1 = (seg1 & 1) << 4;
  const int t_1  = tBase + row1 - 1;
  const bool own1 = tid < 4;
  const bool ok1 = own1 && ((unsigned)t_1 < (unsigned)TT);
  const _Float16* src1 = in_h + (size_t)(bb * TT + t_1) * HD + off1;

  // zero out-of-range rows once (both buffers)
  {
    v8h z = {};
    if (!ok0) {
      _Float16* d0 = &As[0][row0 * LDA + off0];
      _Float16* d1 = &As[1][row0 * LDA + off0];
      *(v8h*)d0 = z; *(v8h*)(d0 + 8) = z;
      *(v8h*)d1 = z; *(v8h*)(d1 + 8) = z;
    }
    if (own1 && !ok1) {
      _Float16* d0 = &As[0][row1 * LDA + off1];
      _Float16* d1 = &As[1][row1 * LDA + off1];
      *(v8h*)d0 = z; *(v8h*)(d0 + 8) = z;
      *(v8h*)d1 = z; *(v8h*)(d1 + 8) = z;
    }
  }

  const int fragOff = (mstripe + l15) * LDA + kh * 8;
  const _Float16* wtLane = wt + ((size_t)(nTile * 8) * 32 + lane) * 16;

  v8f acc[8];
#pragma unroll
  for (int s = 0; s < 8; ++s) acc[s] = (v8f){};

  // ---- stage channel-chunk 0 into buffer 0 ----
  if (ok0) {
    _Float16* d = &As[0][row0 * LDA + off0];
    *(v8h*)d       = *(const v8h*)src0;
    *(v8h*)(d + 8) = *(const v8h*)(src0 + 8);
  }
  if (ok1) {
    _Float16* d = &As[0][row1 * LDA + off1];
    *(v8h*)d       = *(const v8h*)src1;
    *(v8h*)(d + 8) = *(const v8h*)(src1 + 8);
  }

  // ---- prime the B pipeline: (kc=0, tap0, s0..3) ----
  Frag B0[4], B1[4];
  load_group(B0, wtLane);
  __syncthreads();

  for (int kc = 0; kc < NCK; ++kc) {
    const int cur = kc & 1;
    // ---- stage next channel chunk into the other buffer ----
    if (kc + 1 < NCK) {
      const int c0 = (kc + 1) << 5;
      if (ok0) {
        _Float16* d = &As[cur ^ 1][row0 * LDA + off0];
        *(v8h*)d       = *(const v8h*)(src0 + c0);
        *(v8h*)(d + 8) = *(const v8h*)(src0 + c0 + 8);
      }
      if (ok1) {
        _Float16* d = &As[cur ^ 1][row1 * LDA + off1];
        *(v8h*)d       = *(const v8h*)(src1 + c0);
        *(v8h*)(d + 8) = *(const v8h*)(src1 + c0 + 8);
      }
    }
    const _Float16* Acur = As[cur];
    const _Float16* wkc = wtLane + (size_t)kc * CHUNK_H;
    const _Float16* wnx = wtLane + (size_t)(kc + 1 < NCK ? kc + 1 : kc) * CHUNK_H;

    // ---- tap 0 ----
    v16h a = load_afrag(&Acur[fragOff]);
    load_group(B1, wkc + 2048);                 // tap0, s4..7
    wmma4(acc, a, B0);
    load_group(B0, wkc + TAP_H);                // tap1, s0..3
    wmma4(acc + 4, a, B1);
    // ---- tap 1 ----
    a = load_afrag(&Acur[fragOff + LDA]);
    load_group(B1, wkc + TAP_H + 2048);         // tap1, s4..7
    wmma4(acc, a, B0);
    load_group(B0, wkc + 2 * TAP_H);            // tap2, s0..3
    wmma4(acc + 4, a, B1);
    // ---- tap 2 ----
    a = load_afrag(&Acur[fragOff + 2 * LDA]);
    load_group(B1, wkc + 2 * TAP_H + 2048);     // tap2, s4..7
    wmma4(acc, a, B0);
    load_group(B0, wnx);                        // next chunk, tap0, s0..3
    wmma4(acc + 4, a, B1);

    __syncthreads();
  }

  // ---- epilogue: C elem v -> M = kh*8 + v, N = l15; fuse bias ----
  const int mRow = mBase + mstripe + kh * 8;
  const int ocB  = nTile * BN + l15;
#pragma unroll
  for (int s = 0; s < 8; ++s) {
    const int oc = ocB + s * 16;
    const float bv = bias[oc];
#pragma unroll
    for (int v = 0; v < 8; ++v)
      out[(size_t)(mRow + v) * HD + oc] = acc[s][v] + bv;
  }
}

// ---------------------------------------------------------------------------
// y = LayerNorm(relu(h)) * g + b per row (H=512). f32 in, f16 out.
// ---------------------------------------------------------------------------
__global__ __launch_bounds__(256) void relu_ln_kernel(
    const float* __restrict__ h, const float* __restrict__ gam,
    const float* __restrict__ bet, _Float16* __restrict__ out) {
  __shared__ float rs[8], rq[8], bc[2];
  const int row = blockIdx.x;
  const int tid = threadIdx.x;
  const float* hr = h + (size_t)row * HD;
  float x0 = fmaxf(hr[tid], 0.f);
  float x1 = fmaxf(hr[tid + 256], 0.f);
  float s = x0 + x1, q = x0 * x0 + x1 * x1;
#pragma unroll
  for (int off = 16; off > 0; off >>= 1) {
    s += __shfl_down(s, off, 32);
    q += __shfl_down(q, off, 32);
  }
  if ((tid & 31) == 0) { rs[tid >> 5] = s; rq[tid >> 5] = q; }
  __syncthreads();
  if (tid == 0) {
    float ts = 0.f, tq = 0.f;
#pragma unroll
    for (int i = 0; i < 8; ++i) { ts += rs[i]; tq += rq[i]; }
    float mean = ts * (1.f / HD);
    float var  = tq * (1.f / HD) - mean * mean;
    bc[0] = mean;
    bc[1] = rsqrtf(var + EPSF);
  }
  __syncthreads();
  const float mean = bc[0], inv = bc[1];
  _Float16* orow = out + (size_t)row * HD;
  orow[tid]       = (_Float16)((x0 - mean) * inv * gam[tid]       + bet[tid]);
  orow[tid + 256] = (_Float16)((x1 - mean) * inv * gam[tid + 256] + bet[tid + 256]);
}

// ---------------------------------------------------------------------------
// pred[row] = mask ? 0 : dot(h[row], lw) + lb   (h is f16)
// ---------------------------------------------------------------------------
__global__ __launch_bounds__(256) void linear_mask_kernel(
    const _Float16* __restrict__ h, const float* __restrict__ lw,
    const float* __restrict__ lb, const unsigned char* __restrict__ mask,
    float* __restrict__ pred) {
  __shared__ float rs[8];
  const int row = blockIdx.x;
  const int tid = threadIdx.x;
  const _Float16* hr = h + (size_t)row * HD;
  float s = (float)hr[tid] * lw[tid] + (float)hr[tid + 256] * lw[tid + 256];
#pragma unroll
  for (int off = 16; off > 0; off >>= 1) s += __shfl_down(s, off, 32);
  if ((tid & 31) == 0) rs[tid >> 5] = s;
  __syncthreads();
  if (tid == 0) {
    float ts = 0.f;
#pragma unroll
    for (int i = 0; i < 8; ++i) ts += rs[i];
    pred[row] = mask[row] ? 0.f : (ts + lb[0]);
  }
}

// ---------------------------------------------------------------------------
// x2 = x + pitch_emb[bucket(pitch)] + energy_emb[bucket(energy)]
// ---------------------------------------------------------------------------
__device__ __forceinline__ int lower_bound_dev(const float* __restrict__ bins,
                                               int n, float v) {
  int lo = 0, hi = n;
  while (lo < hi) {
    int mid = (lo + hi) >> 1;
    if (bins[mid] < v) lo = mid + 1; else hi = mid;
  }
  return lo;
}

__global__ __launch_bounds__(256) void add_var_emb_kernel(
    const float* __restrict__ x, const float* __restrict__ pt,
    const float* __restrict__ et, const float* __restrict__ pbins,
    const float* __restrict__ ebins, const float* __restrict__ pemb,
    const float* __restrict__ eemb, float* __restrict__ x2) {
  const int row = blockIdx.x;
  const int pi = lower_bound_dev(pbins, NBINS - 1, pt[row]);
  const int ei = lower_bound_dev(ebins, NBINS - 1, et[row]);
  const float* xr = x + (size_t)row * HD;
  const float* pr = pemb + (size_t)pi * HD;
  const float* er = eemb + (size_t)ei * HD;
  float* o = x2 + (size_t)row * HD;
  int c = threadIdx.x;
  o[c]       = xr[c]       + pr[c]       + er[c];
  o[c + 256] = xr[c + 256] + pr[c + 256] + er[c + 256];
}

// ---------------------------------------------------------------------------
// Per-batch inclusive cumsum of durations (T=512) + mel_len.
// ---------------------------------------------------------------------------
__global__ __launch_bounds__(512) void cumsum_kernel(
    const int* __restrict__ dur, int* __restrict__ csum,
    float* __restrict__ mel_len) {
  __shared__ int s[TT];
  const int b = blockIdx.x, t = threadIdx.x;
  s[t] = dur[b * TT + t];
  __syncthreads();
#pragma unroll
  for (int off = 1; off < TT; off <<= 1) {
    int v = (t >= off) ? s[t - off] : 0;
    __syncthreads();
    s[t] += v;
    __syncthreads();
  }
  csum[b * TT + t] = s[t];
  if (t == TT - 1) mel_len[b] = (float)min(s[TT - 1], LMAX);
}

// ---------------------------------------------------------------------------
// Length regulate: out[b,l,:] = x2[b, idx(b,l), :] * (l < mel_len[b])
// ---------------------------------------------------------------------------
__global__ __launch_bounds__(128) void regulate_kernel(
    const float* __restrict__ x2, const int* __restrict__ csum,
    float* __restrict__ out) {
  const int bl = blockIdx.x;
  const int b = bl / LMAX, l = bl % LMAX;
  const int* c = csum + b * TT;
  int lo = 0, hi = TT;
  while (lo < hi) {
    int mid = (lo + hi) >> 1;
    if (c[mid] <= l) lo = mid + 1; else hi = mid;
  }
  const int idx = min(lo, TT - 1);
  const int mel = min(c[TT - 1], LMAX);
  const float sc = (l < mel) ? 1.f : 0.f;
  const float4* src = (const float4*)(x2 + ((size_t)b * TT + idx) * HD);
  float4* dst = (float4*)(out + ((size_t)b * LMAX + l) * HD);
  float4 v = src[threadIdx.x];
  dst[threadIdx.x] = make_float4(v.x * sc, v.y * sc, v.z * sc, v.w * sc);
}

// ---------------------------------------------------------------------------
extern "C" void kernel_launch(void* const* d_in, const int* in_sizes, int n_in,
                              void* d_out, int out_size, void* d_ws,
                              size_t ws_size, hipStream_t stream) {
  (void)in_sizes; (void)n_in; (void)out_size; (void)ws_size;

  const float* x          = (const float*)d_in[0];
  const unsigned char* sm = (const unsigned char*)d_in[1];
  const int*   dur        = (const int*)d_in[2];
  const float* pt         = (const float*)d_in[3];
  const float* et         = (const float*)d_in[4];
  const float* pbins      = (const float*)d_in[24];
  const float* ebins      = (const float*)d_in[25];
  const float* pemb       = (const float*)d_in[26];
  const float* eemb       = (const float*)d_in[27];

  float* out = (float*)d_out;
  const size_t oPitch = (size_t)BATCH * LMAX * HD;
  const size_t oEn    = oPitch + (size_t)BATCH * TT;
  const size_t oLogD  = oEn + (size_t)BATCH * TT;
  const size_t oMel   = oLogD + (size_t)BATCH * TT;

  // workspace carve-up
  char* w = (char*)d_ws;
  _Float16* xh  = (_Float16*)w; w += (size_t)BATCH * TT * HD * sizeof(_Float16);
  _Float16* wt  = (_Float16*)w; w += (size_t)NLAY * WT_LAYER * sizeof(_Float16);
  float*    h0  = (float*)w;    w += (size_t)BATCH * TT * HD * sizeof(float);
  _Float16* h1h = (_Float16*)w; w += (size_t)BATCH * TT * HD * sizeof(_Float16);
  int*      csum = (int*)w;     w += (size_t)BATCH * TT * sizeof(int);
  float*    x2  = h0;  // reused after predictors finish

  const size_t nElem = (size_t)BATCH * TT * HD;
  f32_to_f16_kernel<<<nElem / 1024, 256, 0, stream>>>(x, xh);

  const dim3 gGemm((BATCH * TT) / BM, HD / BN);     // (128, 4)
  const size_t predOut[3] = {oLogD, oPitch, oEn};   // dur, pit, en

  for (int p = 0; p < 3; ++p) {
    const int ib = 6 + p * 6;
    const float* cw = (const float*)d_in[ib + 0];
    const float* cb = (const float*)d_in[ib + 1];
    const float* lg = (const float*)d_in[ib + 2];
    const float* lb = (const float*)d_in[ib + 3];
    const float* lw = (const float*)d_in[ib + 4];
    const float* lbias = (const float*)d_in[ib + 5];

    repack_w_kernel<<<dim3(WT_LAYER / 256, NLAY), 256, 0, stream>>>(cw, wt);
    for (int lay = 0; lay < NLAY; ++lay) {
      const _Float16* in_ptr = (lay == 0) ? xh : h1h;
      conv_gemm_kernel<<<gGemm, 256, 0, stream>>>(
          in_ptr, wt + (size_t)lay * WT_LAYER, cb + lay * HD, h0);
      relu_ln_kernel<<<BATCH * TT, 256, 0, stream>>>(h0, lg + lay * HD,
                                                     lb + lay * HD, h1h);
    }
    linear_mask_kernel<<<BATCH * TT, 256, 0, stream>>>(h1h, lw, lbias, sm,
                                                       out + predOut[p]);
  }

  add_var_emb_kernel<<<BATCH * TT, 256, 0, stream>>>(x, pt, et, pbins, ebins,
                                                     pemb, eemb, x2);
  cumsum_kernel<<<BATCH, 512, 0, stream>>>(dur, csum, out + oMel);
  regulate_kernel<<<BATCH * LMAX, 128, 0, stream>>>(x2, csum, out);
}
